// SpatialCrossAttention_64020782514307
// MI455X (gfx1250) — matrix operand order repaired and checked by reading, not compiled, not run on hardware
//
#include <hip/hip_runtime.h>
#include <hip/hip_bf16.h>

// ---------------------------------------------------------------------------
// Types for CDNA5 WMMA (wave32): A/B = v16h (16 halves = 8 VGPRs), C/D = v8f.
// ---------------------------------------------------------------------------
typedef __attribute__((ext_vector_type(16))) _Float16 v16h;
typedef __attribute__((ext_vector_type(8)))  _Float16 v8h;
typedef __attribute__((ext_vector_type(4)))  _Float16 v4h;
typedef __attribute__((ext_vector_type(8)))  float    v8f;
typedef __attribute__((ext_vector_type(4)))  float    v4f;
typedef __attribute__((ext_vector_type(4)))  int      v4i;

union V16U { v16h v; v8h h[2]; };

// CDNA5 async global->LDS copy (ASYNCcnt) if the toolchain exposes it.
#if defined(__gfx1250__)
#if __has_builtin(__builtin_amdgcn_global_load_async_to_lds_b128) && \
    __has_builtin(__builtin_amdgcn_s_wait_asynccnt)
#define USE_ASYNC_COPY 1
#endif
#if __has_builtin(__builtin_amdgcn_update_dpp)
#define USE_DPP_XOR 1
#endif
#endif

static __device__ __forceinline__ void copy16_g2l(const _Float16* g, _Float16* l) {
#ifdef USE_ASYNC_COPY
  // Builtin signature (from compiler diagnostic): (v4i AS1*, v4i AS3*, Ii, Ii)
  __builtin_amdgcn_global_load_async_to_lds_b128(
      (__attribute__((address_space(1))) v4i*)(void*)g,
      (__attribute__((address_space(3))) v4i*)(void*)l, 0, 0);
#else
  *(v8h*)l = *(const v8h*)g;
#endif
}

static __device__ __forceinline__ void wait_async_copies() {
#ifdef USE_ASYNC_COPY
  __builtin_amdgcn_s_wait_asynccnt(0);
#endif
}

// Butterfly max across the 16-lane group (lane ^ MSK, MSK in {1,2,4,8}).
// DPP16 ROW_XMASK (0x160|MSK) keeps it in the VALU; fallback = ds_bpermute.
template <int MSK>
static __device__ __forceinline__ float xor16_max_step(float x) {
#ifdef USE_DPP_XOR
  int v = __builtin_amdgcn_update_dpp(0, __builtin_bit_cast(int, x),
                                      0x160 | MSK, 0xf, 0xf, true);
  return fmaxf(x, __builtin_bit_cast(float, v));
#else
  return fmaxf(x, __shfl_xor(x, MSK, 32));
#endif
}

static __device__ __forceinline__ v8f wmma_f16(v16h a, v16h b, v8f c) {
  // D = A(16x32 f16) * B(32x16 f16) + C(16x16 f32)
  return __builtin_amdgcn_wmma_f32_16x16x32_f16(false, a, false, b, (short)0, c,
                                                false, false);
}

static __device__ __forceinline__ v8f v8f_zero() {
  v8f z; for (int i = 0; i < 8; ++i) z[i] = 0.0f; return z;
}

// A-matrix fragment 16x32 f16 from row-major storage (stride in halves).
// Lane m=l&15 holds row m; lane[4]=hi selects K-halves:
//   elems 0..7  -> K = kb + 8*hi + 0..7 ; elems 8..15 -> K = kb + 16 + 8*hi + 0..7
static __device__ __forceinline__ v16h load_a_frag(const _Float16* p, int stride,
                                                   int kb, int l15, int hi) {
  V16U r;
  const _Float16* row = p + (size_t)l15 * stride;
  r.h[0] = *(const v8h*)(row + kb + 8 * hi);
  r.h[1] = *(const v8h*)(row + kb + 16 + 8 * hi);
  return r.v;
}

// B-matrix fragment 32x16 f16 from N-major storage: row = N, contiguous K.
// Lane n=l&15 holds column n; elems 0..7 -> K = kb + 8*hi + 0..7 ; elems 8..15 -> K = kb + 16 + 8*hi + 0..7
static __device__ __forceinline__ v16h load_b_frag(const _Float16* p, int stride,
                                                   int kb, int l15, int hi) {
  V16U r;
  const _Float16* row = p + (size_t)l15 * stride;
  r.h[0] = *(const v8h*)(row + kb + 8 * hi);
  r.h[1] = *(const v8h*)(row + kb + 16 + 8 * hi);
  return r.v;
}

// ---------------------------------------------------------------------------
// Problem constants
// ---------------------------------------------------------------------------
#define DMODEL 1024
#define NHEADS 16
#define HDIM   64
#define SEQQ   2048
#define SEQKV  2048
#define BATCH  2
#define MTOT   (BATCH * SEQQ)   // 4096

// GEMM tiling
#define BM 128
#define BN 128
#define BK 64
#define LDK 72   // BK + 8 halves -> 144B row pitch (16B aligned, conflict-spread)

// ---------------------------------------------------------------------------
// Prep: f32 -> f16 elementwise (activations)
// ---------------------------------------------------------------------------
__global__ __launch_bounds__(256) void cvt_f16_kernel(const float* __restrict__ in,
                                                      _Float16* __restrict__ out) {
  const int i = (blockIdx.x * 256 + threadIdx.x) * 4;
  v4f d = *(const v4f*)(in + i);
  v4h h;
  for (int j = 0; j < 4; ++j) h[j] = (_Float16)d[j];
  *(v4h*)(out + i) = h;
}

// Prep: W[k][n] f32 -> Wt[n][k] f16 (32x32 tiles through LDS)
__global__ void wtrans_kernel(const float* __restrict__ W,
                              _Float16* __restrict__ Wt) {
  __shared__ _Float16 s[32][33];
  const int x = threadIdx.x, y = threadIdx.y;
  const int n0 = blockIdx.x * 32, k0 = blockIdx.y * 32;
  for (int i = y; i < 32; i += 8)
    s[i][x] = (_Float16)W[(size_t)(k0 + i) * DMODEL + n0 + x];  // s[k][n]
  __syncthreads();
  for (int i = y; i < 32; i += 8)
    Wt[(size_t)(n0 + i) * DMODEL + k0 + x] = s[x][i];           // Wt[n][k]
}

// ---------------------------------------------------------------------------
// Projection GEMM: C[M,N] = A[M,K] * W[K,N] (+bias); A f16 [M][K], Wt f16 [N][K].
// Double-buffered LDS staging (async global->LDS when available), BK=64,
// 8 waves (4Mx2N), 2x4 WMMA tiles per wave per k-step => 16 WMMA / iter.
// MODE 0: f16 Q[b,h,s,hd] * 1/sqrt(HDIM)   MODE 1: f16 K[b,h,s,hd]
// MODE 2: f16 Vt[b,h,hd,s]                 MODE 3: f32 x = C + bias + residual
// ---------------------------------------------------------------------------
template <int MODE>
__global__ __launch_bounds__(256) void proj_gemm_kernel(
    const _Float16* __restrict__ A, const _Float16* __restrict__ Wt,
    const float* __restrict__ bias, const float* __restrict__ residual,
    void* __restrict__ outp) {
  __shared__ _Float16 As[2][BM * LDK];
  __shared__ _Float16 Bs[2][BN * LDK];

  const int tid  = threadIdx.x;
  const int wid  = tid >> 5;
  const int lane = tid & 31;
  const int l15  = lane & 15;
  const int hi   = lane >> 4;
  const int wm   = wid & 3;     // 4 waves along M
  const int wn   = wid >> 2;    // 2 waves along N
  const int bn0  = blockIdx.x * BN;
  const int bm0  = blockIdx.y * BM;

  v8f acc[2][4];
  for (int i = 0; i < 2; ++i)
    for (int j = 0; j < 4; ++j) acc[i][j] = v8f_zero();

  // staging: 2 threads per row, 32 halves (4x b128) each
  const int srow = tid >> 1;
  const int sseg = (tid & 1) * 32;
  const _Float16* ag = A  + (size_t)(bm0 + srow) * DMODEL + sseg;
  const _Float16* bg = Wt + (size_t)(bn0 + srow) * DMODEL + sseg;

  const int NK = DMODEL / BK;  // 16

  // prologue: stage slab 0 into buffer 0
  for (int j = 0; j < 4; ++j) {
    copy16_g2l(ag + 8 * j, &As[0][srow * LDK + sseg + 8 * j]);
    copy16_g2l(bg + 8 * j, &Bs[0][srow * LDK + sseg + 8 * j]);
  }

  for (int i = 0; i < NK; ++i) {
    const int cur = i & 1;
    wait_async_copies();
    __syncthreads();                       // staged slab visible; prev reads done
    if (i + 1 < NK) {
      const int k0 = (i + 1) * BK;
      for (int j = 0; j < 4; ++j) {
        copy16_g2l(ag + k0 + 8 * j, &As[cur ^ 1][srow * LDK + sseg + 8 * j]);
        copy16_g2l(bg + k0 + 8 * j, &Bs[cur ^ 1][srow * LDK + sseg + 8 * j]);
      }
    }
    // compute on current slab: two k-steps of 32
    for (int ks = 0; ks < BK; ks += 32) {
      v16h af[2], bf[4];
      for (int tm = 0; tm < 2; ++tm)
        af[tm] = load_a_frag(&As[cur][(wm * 32 + tm * 16) * LDK], LDK, ks, l15, hi);
      for (int tn = 0; tn < 4; ++tn)
        bf[tn] = load_b_frag(&Bs[cur][(wn * 64 + tn * 16) * LDK], LDK, ks, l15, hi);
      for (int tm = 0; tm < 2; ++tm)
        for (int tn = 0; tn < 4; ++tn)
          acc[tm][tn] = wmma_f16(af[tm], bf[tn], acc[tm][tn]);
    }
  }

  // epilogue: C/D layout -> lane: n = n0 + (l&15); reg r: m = m0 + r + 8*hi
  for (int tm = 0; tm < 2; ++tm) {
    for (int tn = 0; tn < 4; ++tn) {
      const int n = bn0 + wn * 64 + tn * 16 + l15;
      const float bv = bias[n];
      for (int r = 0; r < 8; ++r) {
        const int m = bm0 + wm * 32 + tm * 16 + r + 8 * hi;
        float v = acc[tm][tn][r] + bv;
        const int b  = m >> 11;          // / SEQQ
        const int s  = m & (SEQQ - 1);
        const int h  = n >> 6;           // / HDIM
        const int hd = n & (HDIM - 1);
        if constexpr (MODE == 0) {
          ((_Float16*)outp)[(((size_t)b * NHEADS + h) * SEQQ + s) * HDIM + hd] =
              (_Float16)(v * 0.125f);    // fold 1/sqrt(64)
        } else if constexpr (MODE == 1) {
          ((_Float16*)outp)[(((size_t)b * NHEADS + h) * SEQKV + s) * HDIM + hd] =
              (_Float16)v;
        } else if constexpr (MODE == 2) {
          ((_Float16*)outp)[(((size_t)b * NHEADS + h) * HDIM + hd) * SEQKV + s] =
              (_Float16)v;
        } else {
          ((float*)outp)[(size_t)m * DMODEL + n] =
              v + residual[(size_t)m * DMODEL + n];
        }
      }
    }
  }
  (void)residual;
}

// ---------------------------------------------------------------------------
// Flash attention: grid (SEQQ/64, B*H), 4 waves x 16 query rows each.
// Q,K: [b,h,s,hd] f16 ; Vt: [b,h,hd,s] f16 ; out attO: [b,s,d] f16.
// waves_per_eu(1): allow high VGPR count so all 16 B-fragments (K and V) can
// stay in flight across the WMMA bursts instead of serializing on loadcnt.
// Row-sum of P via WMMA against ones; row-max via DPP ROW_XMASK butterflies.
// ---------------------------------------------------------------------------
__global__ __launch_bounds__(128)
__attribute__((amdgpu_waves_per_eu(1)))
void flash_attn_kernel(
    const _Float16* __restrict__ Qh, const _Float16* __restrict__ Kh,
    const _Float16* __restrict__ Vt, _Float16* __restrict__ attO) {
  __shared__ _Float16 Plds[4][16 * 72];  // per-wave 16x64 P tile, pitch 72

  const int tid  = threadIdx.x;
  const int wid  = tid >> 5;
  const int lane = tid & 31;
  const int l15  = lane & 15;
  const int hi   = lane >> 4;
  const int q0   = blockIdx.x * 64 + wid * 16;
  const int bh   = blockIdx.y;             // b*NHEADS + h
  const int b    = bh >> 4;
  const int h    = bh & (NHEADS - 1);

  const _Float16* Qbase = Qh + ((size_t)bh * SEQQ + q0) * HDIM;
  const _Float16* Kbase = Kh + (size_t)bh * SEQKV * HDIM;
  const _Float16* Vbase = Vt + (size_t)bh * HDIM * SEQKV;
  _Float16* Pw = Plds[wid];

  v16h aq[2];
  aq[0] = load_a_frag(Qbase, HDIM, 0, l15, hi);
  aq[1] = load_a_frag(Qbase, HDIM, 32, l15, hi);

  v16h onesf;
  for (int i = 0; i < 16; ++i) onesf[i] = (_Float16)1.0f;

  float rowmax[8], rowsum[8];
  v8f oacc[4];
  for (int r = 0; r < 8; ++r) { rowmax[r] = -3.0e38f; rowsum[r] = 0.0f; }
  for (int t = 0; t < 4; ++t) oacc[t] = v8f_zero();

  for (int kc = 0; kc < SEQKV; kc += 64) {
    // prefetch next chunk of K rows and Vt row-segments
    if (kc + 64 < SEQKV) {
      __builtin_prefetch(Kbase + (size_t)(kc + 64 + lane) * HDIM, 0, 3);
      __builtin_prefetch(Kbase + (size_t)(kc + 96 + lane) * HDIM, 0, 3);
      __builtin_prefetch(Vbase + (size_t)(lane * 2) * SEQKV + kc + 64, 0, 3);
      __builtin_prefetch(Vbase + (size_t)(lane * 2 + 1) * SEQKV + kc + 64, 0, 3);
    }

    // ---- issue ALL loads for this chunk: 8 K B-frags + 8 V B-frags ----
    v16h bk[8];
    for (int t = 0; t < 4; ++t) {
      const _Float16* kp = Kbase + (size_t)(kc + t * 16) * HDIM;
      bk[t * 2 + 0] = load_b_frag(kp, HDIM, 0, l15, hi);
      bk[t * 2 + 1] = load_b_frag(kp, HDIM, 32, l15, hi);
    }
    v16h bv[8];
    for (int t = 0; t < 4; ++t) {
      const _Float16* vp = Vbase + (size_t)(t * 16) * SEQKV;
      bv[t * 2 + 0] = load_b_frag(vp, SEQKV, kc, l15, hi);
      bv[t * 2 + 1] = load_b_frag(vp, SEQKV, kc + 32, l15, hi);
    }

    // ---- S = Q * K^T (Q pre-scaled): 8 WMMAs ----
    v8f sacc[4];
    for (int t = 0; t < 4; ++t) {
      sacc[t] = wmma_f16(aq[0], bk[t * 2 + 0], v8f_zero());
      sacc[t] = wmma_f16(aq[1], bk[t * 2 + 1], sacc[t]);
    }

    // ---- online softmax: row-max across 16 lanes (DPP butterfly) ----
    float curmax[8];
    for (int r = 0; r < 8; ++r) {
      float m0 = fmaxf(fmaxf(sacc[0][r], sacc[1][r]),
                       fmaxf(sacc[2][r], sacc[3][r]));
      m0 = xor16_max_step<1>(m0);
      m0 = xor16_max_step<2>(m0);
      m0 = xor16_max_step<4>(m0);
      m0 = xor16_max_step<8>(m0);
      curmax[r] = m0;
    }

    float alpha[8];
    for (int r = 0; r < 8; ++r) {
      float nm = fmaxf(rowmax[r], curmax[r]);
      alpha[r] = __expf(rowmax[r] - nm);
      rowmax[r] = nm;
    }
    for (int t = 0; t < 4; ++t)
      for (int r = 0; r < 8; ++r)
        sacc[t][r] = __expf(sacc[t][r] - rowmax[r]);
    for (int t = 0; t < 4; ++t)
      for (int r = 0; r < 8; ++r) oacc[t][r] *= alpha[r];

    // ---- re-swizzle P: C-layout regs -> LDS row-major (per-wave, in-order) ----
    for (int t = 0; t < 4; ++t)
      for (int r = 0; r < 8; ++r)
        Pw[(r + 8 * hi) * 72 + t * 16 + l15] = (_Float16)sacc[t][r];

    // P A-fragments, reused for row-sum WMMA and for P*V
    v16h pa[2];
    pa[0] = load_a_frag(Pw, 72, 0, l15, hi);
    pa[1] = load_a_frag(Pw, 72, 32, l15, hi);

    // row-sum via WMMA against ones: every column of cs = sum_k P[m][k]
    v8f cs = wmma_f16(pa[0], onesf, v8f_zero());
    cs = wmma_f16(pa[1], onesf, cs);
    for (int r = 0; r < 8; ++r) rowsum[r] = rowsum[r] * alpha[r] + cs[r];

    // ---- O += P * V ----
    for (int t = 0; t < 4; ++t) {
      oacc[t] = wmma_f16(pa[0], bv[t * 2 + 0], oacc[t]);
      oacc[t] = wmma_f16(pa[1], bv[t * 2 + 1], oacc[t]);
    }
  }

  // ---- normalize and write attended [b, s, d] as f16 ----
  float inv[8];
  for (int r = 0; r < 8; ++r) inv[r] = 1.0f / rowsum[r];
  for (int t = 0; t < 4; ++t)
    for (int r = 0; r < 8; ++r) {
      const int q = q0 + r + 8 * hi;
      const int d = h * HDIM + t * 16 + l15;
      attO[((size_t)b * SEQQ + q) * DMODEL + d] = (_Float16)(oacc[t][r] * inv[r]);
    }
}

// ---------------------------------------------------------------------------
// Row LayerNorm over 1024 elements: one block per row, 256 threads x 4 elems.
// ---------------------------------------------------------------------------
__global__ __launch_bounds__(256) void layernorm_kernel(
    const float* __restrict__ x, const float* __restrict__ gamma,
    const float* __restrict__ beta, float* __restrict__ out) {
  __shared__ float red[16];
  const int tid = threadIdx.x, wid = tid >> 5, lane = tid & 31;
  const size_t row = blockIdx.x;
  const int base = tid * 4;
  const float* xr = x + row * DMODEL;

  v4f d = *(const v4f*)(xr + base);
  float s = 0.0f, s2 = 0.0f;
  for (int i = 0; i < 4; ++i) { s += d[i]; s2 += d[i] * d[i]; }
  for (int msk = 16; msk >= 1; msk >>= 1) {
    s += __shfl_xor(s, msk, 32);
    s2 += __shfl_xor(s2, msk, 32);
  }
  if (lane == 0) { red[wid] = s; red[8 + wid] = s2; }
  __syncthreads();
  float ts = 0.0f, ts2 = 0.0f;
  for (int i = 0; i < 8; ++i) { ts += red[i]; ts2 += red[8 + i]; }
  const float mu = ts * (1.0f / DMODEL);
  const float var = ts2 * (1.0f / DMODEL) - mu * mu;
  const float rinv = rsqrtf(var + 1e-5f);

  v4f g = *(const v4f*)(gamma + base);
  v4f be = *(const v4f*)(beta + base);
  v4f o;
  for (int i = 0; i < 4; ++i) o[i] = (d[i] - mu) * rinv * g[i] + be[i];
  *(v4f*)(out + row * DMODEL + base) = o;
}

// ---------------------------------------------------------------------------
extern "C" void kernel_launch(void* const* d_in, const int* in_sizes, int n_in,
                              void* d_out, int out_size, void* d_ws, size_t ws_size,
                              hipStream_t stream) {
  const float* query = (const float*)d_in[0];
  const float* key   = (const float*)d_in[1];
  const float* value = (const float*)d_in[2];
  const float* Wq = (const float*)d_in[3];
  const float* bq = (const float*)d_in[4];
  const float* Wk = (const float*)d_in[5];
  const float* bk = (const float*)d_in[6];
  const float* Wv = (const float*)d_in[7];
  const float* bv = (const float*)d_in[8];
  const float* Wo = (const float*)d_in[9];
  const float* bo = (const float*)d_in[10];
  const float* g  = (const float*)d_in[11];
  const float* be = (const float*)d_in[12];
  (void)in_sizes; (void)n_in; (void)out_size; (void)ws_size;

  char* ws = (char*)d_ws;
  const size_t szTok = (size_t)MTOT * DMODEL * sizeof(_Float16);  // 8 MB
  const size_t szW   = (size_t)DMODEL * DMODEL * sizeof(_Float16);// 2 MB
  const size_t szH   = (size_t)BATCH * NHEADS * SEQKV * HDIM * sizeof(_Float16);

  _Float16* q16 = (_Float16*)(ws + 0 * szTok);
  _Float16* k16 = (_Float16*)(ws + 1 * szTok);
  _Float16* v16 = (_Float16*)(ws + 2 * szTok);
  char* wbase   = ws + 3 * szTok;
  _Float16* Wtq = (_Float16*)(wbase + 0 * szW);
  _Float16* Wtk = (_Float16*)(wbase + 1 * szW);
  _Float16* Wtv = (_Float16*)(wbase + 2 * szW);
  _Float16* Wto = (_Float16*)(wbase + 3 * szW);
  char* hbase   = wbase + 4 * szW;
  _Float16* Qh   = (_Float16*)(hbase + 0 * szH);
  _Float16* Khh  = (_Float16*)(hbase + 1 * szH);
  _Float16* Vt   = (_Float16*)(hbase + 2 * szH);
  _Float16* attO = (_Float16*)(hbase + 3 * szH);
  float*    xbuf = (float*)   (hbase + 4 * szH);

  const int nCvtBlk = (MTOT * DMODEL) / (256 * 4);  // 4096
  cvt_f16_kernel<<<nCvtBlk, 256, 0, stream>>>(query, q16);
  cvt_f16_kernel<<<nCvtBlk, 256, 0, stream>>>(key,   k16);
  cvt_f16_kernel<<<nCvtBlk, 256, 0, stream>>>(value, v16);

  dim3 wtg(DMODEL / 32, DMODEL / 32), wtb(32, 8);
  wtrans_kernel<<<wtg, wtb, 0, stream>>>(Wq, Wtq);
  wtrans_kernel<<<wtg, wtb, 0, stream>>>(Wk, Wtk);
  wtrans_kernel<<<wtg, wtb, 0, stream>>>(Wv, Wtv);
  wtrans_kernel<<<wtg, wtb, 0, stream>>>(Wo, Wto);

  dim3 gg(DMODEL / BN, MTOT / BM);  // (8, 32)
  proj_gemm_kernel<0><<<gg, 256, 0, stream>>>(q16, Wtq, bq, nullptr, Qh);
  proj_gemm_kernel<1><<<gg, 256, 0, stream>>>(k16, Wtk, bk, nullptr, Khh);
  proj_gemm_kernel<2><<<gg, 256, 0, stream>>>(v16, Wtv, bv, nullptr, Vt);

  flash_attn_kernel<<<dim3(SEQQ / 64, BATCH * NHEADS), 128, 0, stream>>>(Qh, Khh, Vt, attO);

  proj_gemm_kernel<3><<<gg, 256, 0, stream>>>(attO, Wto, bo, query, xbuf);
  layernorm_kernel<<<MTOT, 256, 0, stream>>>(xbuf, g, be, (float*)d_out);
}
